// KernelAttention_70497593197054
// MI455X (gfx1250) — matrix operand order repaired and checked by reading, not compiled, not run on hardware
//
#include <hip/hip_runtime.h>
#include <hip/hip_bf16.h>

// ---------------------------------------------------------------------------
// Types for CDNA5 WMMA (wave32)
// ---------------------------------------------------------------------------
typedef __attribute__((ext_vector_type(16))) __bf16        v16bf;
typedef __attribute__((ext_vector_type(8)))  float         v8f;
typedef __attribute__((ext_vector_type(4)))  unsigned int  u32x4;

union FragB16 { v16bf v; u32x4 q[2]; };

// hardware f32 -> bf16 (RNE); falls back to integer expansion if no cvt op
__device__ __forceinline__ unsigned short f2bf(float f) {
  __bf16 h = (__bf16)f;
  return __builtin_bit_cast(unsigned short, h);
}

// ---------------------------------------------------------------------------
// Problem constants
// ---------------------------------------------------------------------------
#define DIMC 64
#define WIN  4
#define LTOK 36       // 4 windows * 3*3 taps
#define HDS  8
#define DH   8
#define RED  4
#define BN   8
#define HH   256
#define WW2  256
#define WSZ  128      // window size
#define KK   576      // 64 in-ch * 9 taps (ordered tap*64 + c)

// ---------------------------------------------------------------------------
// P1: qkv[d][l][j] = sum_i conv_w[w(l)][d][i][tap(l)] * w_qkv[j][i] + b_qkv[j]
// 64*36*192 = 442368 threads (1728 blocks exactly)
// ---------------------------------------------------------------------------
__global__ __launch_bounds__(256) void ka_qkv(
    const float* __restrict__ conv_w, const float* __restrict__ w_qkv,
    const float* __restrict__ b_qkv, float* __restrict__ qkv)
{
  int idx = blockIdx.x * 256 + threadIdx.x;
  int j  = idx % 192;
  int dl = idx / 192;
  int l  = dl % LTOK, d = dl / LTOK;
  int w  = l / 9, tap = l % 9;
  const float* kp = conv_w + (((size_t)w * DIMC + d) * DIMC) * 9 + tap; // stride 9 over i
  const float* wp = w_qkv + (size_t)j * DIMC;
  float s = b_qkv[j];
  #pragma unroll 8
  for (int i = 0; i < DIMC; ++i) s += kp[i * 9] * wp[i];
  qkv[idx] = s;
}

// ---------------------------------------------------------------------------
// P2: per-(d,h,l) attention row: scores over 36 keys, softmax, weighted V sum
// 64*8*36 = 18432 threads (72 blocks exactly)
// qkv layout: [d][l][ {q|k|v}*64 + h*8 + e ]
// ---------------------------------------------------------------------------
__global__ __launch_bounds__(256) void ka_attn(
    const float* __restrict__ qkv, float* __restrict__ ao)
{
  int idx = blockIdx.x * 256 + threadIdx.x;
  int l = idx % LTOK;
  int h = (idx / LTOK) & 7;
  int d = idx / (LTOK * HDS);
  const float* qp = qkv + ((size_t)d * LTOK + l) * 192 + h * DH;
  float sc[LTOK];
  float mx = -3.0e38f;
  for (int m = 0; m < LTOK; ++m) {
    const float* kp = qkv + ((size_t)d * LTOK + m) * 192 + 64 + h * DH;
    float s = 0.f;
    #pragma unroll
    for (int e = 0; e < DH; ++e) s += qp[e] * kp[e];
    s *= 0.35355339059327373f;     // DH^-0.5
    sc[m] = s;
    mx = fmaxf(mx, s);
  }
  float sum = 0.f;
  for (int m = 0; m < LTOK; ++m) { sc[m] = __expf(sc[m] - mx); sum += sc[m]; }
  float inv = 1.f / sum;
  float o[DH] = {0.f, 0.f, 0.f, 0.f, 0.f, 0.f, 0.f, 0.f};
  for (int m = 0; m < LTOK; ++m) {
    const float* vp = qkv + ((size_t)d * LTOK + m) * 192 + 128 + h * DH;
    float pm = sc[m] * inv;
    #pragma unroll
    for (int e = 0; e < DH; ++e) o[e] += pm * vp[e];
  }
  float* op = ao + ((size_t)d * LTOK + l) * DIMC + h * DH;
  #pragma unroll
  for (int e = 0; e < DH; ++e) op[e] = o[e];
}

// ---------------------------------------------------------------------------
// P3: proj[d][l][o] = sum_c ao[d][l][c] * w_out[o][c] + b_out[o]
// 64*36*64 = 147456 threads (576 blocks exactly)
// ---------------------------------------------------------------------------
__global__ __launch_bounds__(256) void ka_proj(
    const float* __restrict__ ao, const float* __restrict__ w_out,
    const float* __restrict__ b_out, float* __restrict__ proj)
{
  int idx = blockIdx.x * 256 + threadIdx.x;
  int o  = idx & 63;
  int dl = idx >> 6;
  const float* ap = ao + (size_t)dl * DIMC;
  const float* wp = w_out + (size_t)o * DIMC;
  float s = b_out[o];
  #pragma unroll 8
  for (int c = 0; c < DIMC; ++c) s += ap[c] * wp[c];
  proj[idx] = s;
}

// ---------------------------------------------------------------------------
// P4: SE gate per (window, out-ch) and emit bf16 weight matrix
//     Wbf[w][d][tap*64 + c] = bf16( proj[d][w*9+tap][c] * sigmoid_gate[w][d][c] )
// 1 block of 256 threads (4 windows * 64 out-ch)
// ---------------------------------------------------------------------------
__global__ __launch_bounds__(256) void ka_se_gate(
    const float* __restrict__ proj,
    const float* __restrict__ se_w1, const float* __restrict__ se_b1,
    const float* __restrict__ se_w2, const float* __restrict__ se_b2,
    unsigned short* __restrict__ Wbf)
{
  int t = threadIdx.x;
  int d = t & 63, w = t >> 6;
  const float* pb = proj + ((size_t)d * LTOK + w * 9) * DIMC;  // [tap][c], tap stride 64
  float hr[RED];
  #pragma unroll
  for (int r = 0; r < RED; ++r) hr[r] = se_b1[w * RED + r];
  for (int c = 0; c < DIMC; ++c) {
    float pc = 0.f;
    #pragma unroll
    for (int tap = 0; tap < 9; ++tap) pc += pb[tap * DIMC + c];
    pc *= (1.f / 9.f);
    #pragma unroll
    for (int r = 0; r < RED; ++r) hr[r] += pc * se_w1[((size_t)w * RED + r) * DIMC + c];
  }
  #pragma unroll
  for (int r = 0; r < RED; ++r) hr[r] = fmaxf(hr[r], 0.f);
  unsigned short* wout = Wbf + ((size_t)w * DIMC + d) * KK;
  for (int c = 0; c < DIMC; ++c) {
    float g = se_b2[w * DIMC + c];
    #pragma unroll
    for (int r = 0; r < RED; ++r) g += hr[r] * se_w2[((size_t)w * DIMC + c) * RED + r];
    float s = 1.f / (1.f + __expf(-g));
    #pragma unroll
    for (int tap = 0; tap < 9; ++tap)
      wout[tap * DIMC + c] = f2bf(pb[tap * DIMC + c] * s);
  }
}

// ---------------------------------------------------------------------------
// Main kernel: dynamic per-window 3x3 conv (implicit GEMM, bf16 WMMA, f32
// accum) fused with channel LayerNorm + residual.
//
// One block = (window w, batch b, 2 output rows, 64-pixel half-row):
//   M = 64 out channels (4 tiles of 16), N = 2 rows x 64 px = 128 pixels,
//   K = 576 = 9 taps * 64 in-ch, stepped 32 at a time (one tap, half the chs).
// 8 waves: wave = mtile*2 + rowsel; each wave -> 16 ch x 64 px of one output
// row = 4 wmma accumulators; one A fragment feeds 4 WMMAs per K step.
// The bf16 halo tile and the f32 y-tile alias the same LDS (never co-live).
// ---------------------------------------------------------------------------
#define SMEM_BYTES (4 * 66 * 72 * 2)   // 38016 B >= 35840 B used in phase 2

__global__ __launch_bounds__(256) void ka_conv_ln(
    const float* __restrict__ x, const unsigned short* __restrict__ Wbf,
    const float* __restrict__ ln_w, const float* __restrict__ ln_b,
    float* __restrict__ out)
{
  __shared__ __align__(16) unsigned char smem[SMEM_BYTES];
  __shared__ float ln_ws[DIMC], ln_bs[DIMC];

  // phase-1 view: 64ch x 4rows x 66cols halo tile, channel-innermost,
  // stride 72 halves (16B-aligned chunks, odd 16B-chunk lane stride for banks)
  unsigned short* lds_x = (unsigned short*)smem;
  // phase-2 view: y[64 ch][128 px] + LN reduction scratch
  float* y_tile = (float*)smem;
  float* red1 = y_tile + DIMC * 128;   // [2][128]
  float* red2 = red1 + 256;            // [2][128]
  float* mu_s = red2 + 256;            // [128]
  float* rs_s = mu_s + 128;            // [128]

  const int blk   = blockIdx.x;        // WIN*B*64*2 = 4096 blocks
  const int xhalf = blk & 1;
  const int py0   = ((blk >> 1) & 63) << 1;   // first of 2 output rows
  const int b     = (blk >> 7) & 7;
  const int widx  = blk >> 10;
  const int wh = widx >> 1, ww = widx & 1;
  const int px0 = xhalf << 6;
  const int tid = threadIdx.x;

  if (tid < DIMC) { ln_ws[tid] = ln_w[tid]; ln_bs[tid] = ln_b[tid]; }

  // ---- stage input halo tile (fp32 global -> bf16 LDS, zero-padded) ----
  // Interior columns (lds col 1..64 -> xx = px0..px0+63) are always in-bounds
  // in x, so only the row needs checking; decode is all shifts and each
  // thread loads a float2 (two adjacent columns) -> coalesced 256B/wave.
  // 4 rows x 32 col-pairs x 64 ch = 8192 pairs = 32 per thread.
  {
    const float* xrow0 = x + ((size_t)b * DIMC) * HH * WW2 + (size_t)(wh * WSZ) * WW2
                           + (ww * WSZ + px0);
    #pragma unroll 4
    for (int e = tid; e < 4 * 32 * DIMC; e += 256) {
      int pr = e & 31;          // column pair index
      int t  = e >> 5;
      int r  = t & 3;
      int c  = t >> 2;
      int yy = py0 + r - 1;     // window-local row
      float2 v = make_float2(0.f, 0.f);
      if ((unsigned)yy < 128u)
        v = *(const float2*)(xrow0 + ((size_t)c * HH + yy) * WW2 + pr * 2);
      unsigned short* dst = &lds_x[(r * 66 + (pr * 2 + 1)) * 72 + c];
      dst[0]  = f2bf(v.x);
      dst[72] = f2bf(v.y);
    }
  }
  // halo columns: lds col 0 (xx = px0-1) and col 65 (xx = px0+64)
  // 4 rows x 2 cols x 64 ch = 512 elements = 2 per thread.
  for (int e = tid; e < 512; e += 256) {
    int hcol = e & 1;
    int t = e >> 1;
    int r = t & 3;
    int c = t >> 2;
    int col = hcol ? 65 : 0;
    int yy = py0 + r - 1;
    int xx = px0 + col - 1;
    float v = 0.f;
    if ((unsigned)yy < 128u && (unsigned)xx < 128u)
      v = x[(((size_t)b * DIMC + c) * HH + (wh * WSZ + yy)) * WW2 + (ww * WSZ + xx)];
    lds_x[(r * 66 + col) * 72 + c] = f2bf(v);
  }
  __syncthreads();

  const int lane   = tid & 31, wave = tid >> 5;
  const int mtile  = wave >> 1;        // 0..3: out-channel tile
  const int rowsel = wave & 1;         // which of the 2 output rows
  const int n  = lane & 15;
  const int hi = lane >> 4;
  const int arow = mtile * 16 + (lane & 15);          // out channel for A frag

  v8f acc[4] = {};
  const unsigned short* wrow = Wbf + ((size_t)widx * DIMC + arow) * KK;

  #pragma unroll
  for (int ks = 0; ks < 18; ++ks) {
    const int tap = ks >> 1;
    const int c0  = (ks & 1) << 5;     // 0 or 32 within this tap's channels
    const int dy  = tap / 3, dx = tap % 3;

    // A frag (16x32 bf16): lanes 0-15 hold K {0..7,16..23}, lanes 16-31 {8..15,24..31}
    FragB16 a;
    const unsigned short* ap = wrow + tap * DIMC + c0 + hi * 8;
    a.q[0] = *(const u32x4*)(ap);
    a.q[1] = *(const u32x4*)(ap + 16);

    // B frags (32x16 bf16): lane = pixel; lanes 0-15 hold K {0..7,16..23},
    // lanes 16-31 K {8..15,24..31} (same order as A)
    const int r_in = rowsel + dy;      // staged input row 0..3
    const unsigned short* bbase = lds_x + (r_in * 66) * 72 + c0 + hi * 8;
    #pragma unroll
    for (int s = 0; s < 4; ++s) {
      FragB16 bf;
      const unsigned short* bp = bbase + (s * 16 + n + dx) * 72;
      bf.q[0] = *(const u32x4*)(bp);
      bf.q[1] = *(const u32x4*)(bp + 16);
      acc[s] = __builtin_amdgcn_wmma_f32_16x16x32_bf16(false, a.v, false, bf.v,
                                                       (short)0, acc[s], false, false);
    }
  }

  // all waves done reading lds_x before y_tile overwrites it
  __syncthreads();

  // ---- scatter C (VGPR j -> M = 8*hi + j, N = lane&15) into y_tile ----
  #pragma unroll
  for (int s = 0; s < 4; ++s) {
    #pragma unroll
    for (int j = 0; j < 8; ++j) {
      int m = mtile * 16 + hi * 8 + j;
      y_tile[m * 128 + rowsel * 64 + s * 16 + n] = acc[s][j];
    }
  }
  __syncthreads();

  // ---- fused channel LayerNorm + residual over 128 pixels ----
  const int p = tid & 127, q = tid >> 7;   // pixel, channel-half
  float s1 = 0.f, s2 = 0.f;
  #pragma unroll 8
  for (int c = q * 32; c < q * 32 + 32; ++c) {
    float v = y_tile[c * 128 + p];
    s1 += v; s2 += v * v;
  }
  red1[q * 128 + p] = s1; red2[q * 128 + p] = s2;
  __syncthreads();
  if (q == 0) {
    float S1 = red1[p] + red1[128 + p];
    float S2 = red2[p] + red2[128 + p];
    float mu  = S1 * (1.f / 64.f);
    float var = S2 * (1.f / 64.f) - mu * mu;
    mu_s[p] = mu;
    rs_s[p] = rsqrtf(var + 1e-5f);
  }
  __syncthreads();
  const float mu = mu_s[p], rs = rs_s[p];
  const int gy = wh * WSZ + py0 + (p >> 6);
  const int gx = ww * WSZ + px0 + (p & 63);
  #pragma unroll 8
  for (int c = q * 32; c < q * 32 + 32; ++c) {
    size_t gi = (((size_t)b * DIMC + c) * HH + gy) * WW2 + gx;
    float yv = (y_tile[c * 128 + p] - mu) * rs * ln_ws[c] + ln_bs[c];
    out[gi] = x[gi] + yv;
  }
}

// ---------------------------------------------------------------------------
// Launch
// ---------------------------------------------------------------------------
extern "C" void kernel_launch(void* const* d_in, const int* in_sizes, int n_in,
                              void* d_out, int out_size, void* d_ws, size_t ws_size,
                              hipStream_t stream) {
  (void)in_sizes; (void)n_in; (void)out_size; (void)ws_size;
  const float* x      = (const float*)d_in[0];
  const float* conv_w = (const float*)d_in[1];
  const float* w_qkv  = (const float*)d_in[2];
  const float* b_qkv  = (const float*)d_in[3];
  const float* w_out  = (const float*)d_in[4];
  const float* b_out  = (const float*)d_in[5];
  const float* se_w1  = (const float*)d_in[6];
  const float* se_b1  = (const float*)d_in[7];
  const float* se_w2  = (const float*)d_in[8];
  const float* se_b2  = (const float*)d_in[9];
  const float* ln_w   = (const float*)d_in[10];
  const float* ln_b   = (const float*)d_in[11];
  float* out = (float*)d_out;

  // workspace layout (fp32 / bf16), ~3.24 MB total
  float* qkv  = (float*)d_ws;                       // 64*36*192 = 442368
  float* ao   = qkv + 64 * LTOK * 192;              // 64*36*64  = 147456
  float* proj = ao  + 64 * LTOK * DIMC;             // 64*36*64  = 147456
  unsigned short* Wbf = (unsigned short*)(proj + 64 * LTOK * DIMC); // 4*64*576

  ka_qkv    <<<1728, 256, 0, stream>>>(conv_w, w_qkv, b_qkv, qkv);
  ka_attn   <<<  72, 256, 0, stream>>>(qkv, ao);
  ka_proj   <<< 576, 256, 0, stream>>>(ao, w_out, b_out, proj);
  ka_se_gate<<<   1, 256, 0, stream>>>(proj, se_w1, se_b1, se_w2, se_b2, Wbf);
  ka_conv_ln<<<4096, 256, 0, stream>>>(x, Wbf, ln_w, ln_b, out);
}
